// SoftMemory_30142080483336
// MI455X (gfx1250) — compile-verified, matrix operand order checked by reading, not run on hardware
//
#include <hip/hip_runtime.h>

#define N_TOKEN  16384
#define N_EMBED  512
#define BM       64     // token rows per workgroup
#define BN       32     // n-block (K of GEMM2, WMMA K for bf16)
#define NTHREADS 256    // 8 waves (wave32)

typedef __attribute__((ext_vector_type(16))) __bf16       v16bf;
typedef __attribute__((ext_vector_type(8)))  float        v8f;
typedef __attribute__((ext_vector_type(4)))  unsigned int u32x4;
typedef __attribute__((ext_vector_type(4)))  int          i32x4;
typedef __attribute__((ext_vector_type(4)))  float        f32x4;

union FragAB { v16bf h; u32x4 q[2]; };

#if defined(__has_builtin)
# if __has_builtin(__builtin_amdgcn_global_load_async_to_lds_b128)
#  define HAVE_ASYNC_LDS 1
# endif
#endif

#ifdef HAVE_ASYNC_LDS
typedef __attribute__((address_space(1))) i32x4 as1_v4i;   // global v4i
typedef __attribute__((address_space(3))) i32x4 as3_v4i;   // LDS v4i
// One 16-byte async global->LDS copy per lane (ASYNCcnt-tracked, no VGPR trip).
__device__ __forceinline__ void cp_async16(void* lds_dst, const void* g_src) {
  as1_v4i* g = (as1_v4i*)(unsigned long long)g_src;
  as3_v4i* l = (as3_v4i*)(unsigned int)(unsigned long long)lds_dst; // LDS offset = addr[31:0]
  __builtin_amdgcn_global_load_async_to_lds_b128(g, l, 0, 0);
}
__device__ __forceinline__ void wait_async0() {
# if __has_builtin(__builtin_amdgcn_s_wait_asynccnt)
  __builtin_amdgcn_s_wait_asynccnt(0);
# else
  asm volatile("s_wait_asynccnt 0x0" ::: "memory");
# endif
}
#endif

__device__ __forceinline__ unsigned short f2bf(float f) {
  unsigned int u = __float_as_uint(f);
  u = (u + 0x7FFFu + ((u >> 16) & 1u)) >> 16;   // round-to-nearest-even bf16
  return (unsigned short)u;
}
__device__ __forceinline__ unsigned int pack2bf(float lo, float hi) {
  return (unsigned int)f2bf(lo) | ((unsigned int)f2bf(hi) << 16);
}

// ---------------------------------------------------------------------------
// One-time prep: lin_w -> bf16 [N,E] (row-major), value_w -> bf16 [E,N]
// (transposed).  Runs once per launch; main loop then does pure b128 copies.
// ---------------------------------------------------------------------------
__global__ __launch_bounds__(NTHREADS)
void prep_bf16(const float* __restrict__ lin_w,
               const float* __restrict__ value_w,
               unsigned short* __restrict__ Wbf,   // [N_TOKEN][N_EMBED]
               unsigned short* __restrict__ Vt)    // [N_EMBED][N_TOKEN]
{
  int t = blockIdx.x * NTHREADS + threadIdx.x;    // one thread per 8 elements
  size_t off = (size_t)t * 8;
  int n = (int)(off >> 9);
  int e = (int)(off & (N_EMBED - 1));

  f32x4 a = *(const f32x4*)(lin_w + off);
  f32x4 b = *(const f32x4*)(lin_w + off + 4);
  u32x4 w;
  w.x = pack2bf(a.x, a.y); w.y = pack2bf(a.z, a.w);
  w.z = pack2bf(b.x, b.y); w.w = pack2bf(b.z, b.w);
  *(u32x4*)(Wbf + off) = w;

  f32x4 c = *(const f32x4*)(value_w + off);
  f32x4 d = *(const f32x4*)(value_w + off + 4);
  Vt[(size_t)(e + 0) * N_TOKEN + n] = f2bf(c.x);
  Vt[(size_t)(e + 1) * N_TOKEN + n] = f2bf(c.y);
  Vt[(size_t)(e + 2) * N_TOKEN + n] = f2bf(c.z);
  Vt[(size_t)(e + 3) * N_TOKEN + n] = f2bf(c.w);
  Vt[(size_t)(e + 4) * N_TOKEN + n] = f2bf(d.x);
  Vt[(size_t)(e + 5) * N_TOKEN + n] = f2bf(d.y);
  Vt[(size_t)(e + 6) * N_TOKEN + n] = f2bf(d.z);
  Vt[(size_t)(e + 7) * N_TOKEN + n] = f2bf(d.w);
}

// ---------------------------------------------------------------------------
// Fused flash-softmax "soft memory":  out = softmax(x Wt + b) V + x
// ---------------------------------------------------------------------------
__global__ __launch_bounds__(NTHREADS, 1)
void softmem_fused(const float* __restrict__ x,
                   const unsigned short* __restrict__ Wbf,  // bf16 [N,E]
                   const float* __restrict__ lin_b,
                   const unsigned short* __restrict__ Vt,   // bf16 [E,N]
                   float* __restrict__ out)
{
  __shared__ unsigned short Xs[BM * N_EMBED];     // 64 KB  bf16 x tile (persistent)
  __shared__ unsigned short Ws[BN * N_EMBED];     // 32 KB  bf16 lin_w block (n,k)
  __shared__ unsigned short VsT[N_EMBED * BN];    // 32 KB  bf16 value block (e,n)
  __shared__ unsigned short Ps[BM * BN];          //  4 KB  bf16 probabilities
  __shared__ float          Sb[BM * BN];          //  8 KB  fp32 logits block
  __shared__ float          mrow[BM], lrow[BM], crow[BM];
  __shared__ int            changed[4];           // per 16-row group rescale flag

  const int tid  = threadIdx.x;
  const int lane = tid & 31;
  const int wave = tid >> 5;
  const int sel  = lane >> 4;       // half-wave selector (K-range per ISA layout)
  const int l15  = lane & 15;
  const int mg   = wave >> 1;       // 0..3 : 16-row group
  const int cg   = wave & 1;        // 0..1 : 256-col group / 16-col S tile
  const int row0 = blockIdx.x * BM;

  // ---- stage x tile -> bf16 in LDS (one-time conversion) ----
  #pragma unroll
  for (int i = 0; i < (BM * N_EMBED) / (NTHREADS * 4); ++i) {
    int idx = (tid + i * NTHREADS) * 4;
    int m = idx >> 9, k = idx & (N_EMBED - 1);
    f32x4 v = *(const f32x4*)(x + (size_t)(row0 + m) * N_EMBED + k);
    unsigned int* dst = (unsigned int*)&Xs[m * N_EMBED + k];
    dst[0] = pack2bf(v.x, v.y);
    dst[1] = pack2bf(v.z, v.w);
  }
  if (tid < BM) { mrow[tid] = -1e30f; lrow[tid] = 0.0f; }

#ifdef HAVE_ASYNC_LDS
  // pre-issue Ws block 0 async
  #pragma unroll
  for (int i = 0; i < 8; ++i) {
    int ci = tid + i * NTHREADS;
    cp_async16((u32x4*)Ws + ci, (const u32x4*)Wbf + ci);
  }
#endif

  v8f acc[16];                       // this wave's 16 x 256 fp32 output strip
  #pragma unroll
  for (int j = 0; j < 16; ++j)
    #pragma unroll
    for (int r = 0; r < 8; ++r) acc[j][r] = 0.0f;

  const float LOG2E = 1.44269504088896340736f;

  for (int n0 = 0; n0 < N_TOKEN; n0 += BN) {
    __syncthreads();   // previous iteration's consumers of VsT/crow are done

    if (tid < 4) changed[tid] = 0;

    // ---- stage weight blocks (bf16, pre-converted) ----
#ifdef HAVE_ASYNC_LDS
    {
      u32x4* vdst = (u32x4*)VsT;     // dst element offset = 8*chunk (linear)
      #pragma unroll
      for (int i = 0; i < 8; ++i) {
        int ci = tid + i * NTHREADS;           // 2048 chunks of 8 elements
        int e = ci >> 2, c = ci & 3;
        cp_async16(vdst + ci, Vt + (size_t)e * N_TOKEN + n0 + c * 8);
      }
      wait_async0();                 // Ws(n0) issued earlier + VsT(n0) complete
    }
#else
    {
      const u32x4* wsrc = (const u32x4*)(Wbf + (size_t)n0 * N_EMBED);
      u32x4*       wdst = (u32x4*)Ws;
      #pragma unroll
      for (int i = 0; i < 8; ++i)
        wdst[tid + i * NTHREADS] = wsrc[tid + i * NTHREADS];
      u32x4* vdst = (u32x4*)VsT;
      #pragma unroll
      for (int i = 0; i < 8; ++i) {
        int ci = tid + i * NTHREADS;
        int e = ci >> 2, c = ci & 3;
        vdst[ci] = *(const u32x4*)(Vt + (size_t)e * N_TOKEN + n0 + c * 8);
      }
    }
#endif
    if (n0 + BN < N_TOKEN) {   // warm GL2 for next V block (W is prefetched async)
      __builtin_prefetch(Vt + (size_t)(tid << 1) * N_TOKEN + n0 + BN, 0, 1);
    }
    __syncthreads();

    // ---- GEMM1: this wave's 16x16 logits tile, K = 512 in 16 WMMA steps ----
    v8f s;
    #pragma unroll
    for (int r = 0; r < 8; ++r) s[r] = 0.0f;
    const unsigned short* arow = &Xs[(mg * 16 + l15) * N_EMBED];
    const unsigned short* brow = &Ws[(cg * 16 + l15) * N_EMBED];
    #pragma unroll
    for (int k0 = 0; k0 < N_EMBED; k0 += 32) {
      FragAB a, b;
      a.q[0] = *(const u32x4*)(arow + k0 + sel * 8);
      a.q[1] = *(const u32x4*)(arow + k0 + 16 + sel * 8);
      b.q[0] = *(const u32x4*)(brow + k0 + sel * 8);
      b.q[1] = *(const u32x4*)(brow + k0 + 16 + sel * 8);
      s = __builtin_amdgcn_wmma_f32_16x16x32_bf16(false, a.h, false, b.h,
                                                  (short)0, s, false, false);
    }
    // bias add + spill S tile (C layout: lanes 0-15 M=r, lanes 16-31 M=8+r)
    float bias = lin_b[n0 + cg * 16 + l15];
    #pragma unroll
    for (int r = 0; r < 8; ++r) {
      int m = mg * 16 + sel * 8 + r;
      Sb[m * BN + cg * 16 + l15] = s[r] + bias;
    }
    __syncthreads();   // all waves done with Ws; Sb visible

#ifdef HAVE_ASYNC_LDS
    // software pipeline: next Ws block streams into LDS during softmax + GEMM2
    if (n0 + BN < N_TOKEN) {
      const u32x4* wsrc = (const u32x4*)(Wbf + (size_t)(n0 + BN) * N_EMBED);
      #pragma unroll
      for (int i = 0; i < 8; ++i) {
        int ci = tid + i * NTHREADS;
        cp_async16((u32x4*)Ws + ci, wsrc + ci);
      }
    }
#endif

    // ---- online softmax: 4 lanes per row, 8 exps each, shfl reductions ----
    {
      int row = tid >> 2, c4 = tid & 3;
      const float* srow = &Sb[row * BN + c4 * 8];
      f32x4 s0 = *(const f32x4*)(srow);
      f32x4 s1 = *(const f32x4*)(srow + 4);
      float mx = fmaxf(fmaxf(fmaxf(s0.x, s0.y), fmaxf(s0.z, s0.w)),
                       fmaxf(fmaxf(s1.x, s1.y), fmaxf(s1.z, s1.w)));
      mx = fmaxf(mx, __shfl_xor(mx, 1, 32));
      mx = fmaxf(mx, __shfl_xor(mx, 2, 32));
      float mo = mrow[row];
      mx = fmaxf(mx, mo);
      float p0 = __builtin_exp2f((s0.x - mx) * LOG2E);
      float p1 = __builtin_exp2f((s0.y - mx) * LOG2E);
      float p2 = __builtin_exp2f((s0.z - mx) * LOG2E);
      float p3 = __builtin_exp2f((s0.w - mx) * LOG2E);
      float p4 = __builtin_exp2f((s1.x - mx) * LOG2E);
      float p5 = __builtin_exp2f((s1.y - mx) * LOG2E);
      float p6 = __builtin_exp2f((s1.z - mx) * LOG2E);
      float p7 = __builtin_exp2f((s1.w - mx) * LOG2E);
      float sum = ((p0 + p1) + (p2 + p3)) + ((p4 + p5) + (p6 + p7));
      sum += __shfl_xor(sum, 1, 32);
      sum += __shfl_xor(sum, 2, 32);
      unsigned int* prow = (unsigned int*)&Ps[row * BN + c4 * 8];
      prow[0] = pack2bf(p0, p1);
      prow[1] = pack2bf(p2, p3);
      prow[2] = pack2bf(p4, p5);
      prow[3] = pack2bf(p6, p7);
      if (c4 == 0) {                       // 4-lane row group is intra-wave:
        float corr = __builtin_exp2f((mo - mx) * LOG2E);
        mrow[row] = mx;                    //  lockstep => reads above precede
        lrow[row] = lrow[row] * corr + sum;
        crow[row] = corr;
        if (!(corr == 1.0f)) changed[row >> 4] = 1;
      }
    }
    __syncthreads();

    // ---- conditional rescale (wave-uniform branch) + GEMM2 ----
    if (changed[mg]) {
      float cf[8];
      #pragma unroll
      for (int r = 0; r < 8; ++r) cf[r] = crow[mg * 16 + sel * 8 + r];
      #pragma unroll
      for (int j = 0; j < 16; ++j)
        #pragma unroll
        for (int r = 0; r < 8; ++r) acc[j][r] *= cf[r];
    }
    FragAB pa;
    const unsigned short* prow = &Ps[(mg * 16 + l15) * BN];
    pa.q[0] = *(const u32x4*)(prow + sel * 8);
    pa.q[1] = *(const u32x4*)(prow + 16 + sel * 8);
    #pragma unroll
    for (int j = 0; j < 16; ++j) {
      FragAB vb;
      const unsigned short* vcol = &VsT[(cg * 256 + j * 16 + l15) * BN];
      vb.q[0] = *(const u32x4*)(vcol + sel * 8);
      vb.q[1] = *(const u32x4*)(vcol + 16 + sel * 8);
      acc[j] = __builtin_amdgcn_wmma_f32_16x16x32_bf16(false, pa.h, false, vb.h,
                                                       (short)0, acc[j], false, false);
    }
  }
  __syncthreads();

  // ---- epilogue: out = acc / l + x (residual in full fp32) ----
  float invl[8];
  #pragma unroll
  for (int r = 0; r < 8; ++r) invl[r] = 1.0f / lrow[mg * 16 + sel * 8 + r];
  #pragma unroll
  for (int j = 0; j < 16; ++j) {
    int e = cg * 256 + j * 16 + l15;
    #pragma unroll
    for (int r = 0; r < 8; ++r) {
      size_t g = (size_t)(row0 + mg * 16 + sel * 8 + r) * N_EMBED + e;
      out[g] = acc[j][r] * invl[r] + x[g];
    }
  }
}

extern "C" void kernel_launch(void* const* d_in, const int* in_sizes, int n_in,
                              void* d_out, int out_size, void* d_ws, size_t ws_size,
                              hipStream_t stream) {
  (void)n_in; (void)out_size; (void)ws_size;
  const float* x       = (const float*)d_in[0];
  const float* lin_w   = (const float*)d_in[1];
  const float* lin_b   = (const float*)d_in[2];
  const float* value_w = (const float*)d_in[3];
  float* out = (float*)d_out;

  unsigned short* Wbf = (unsigned short*)d_ws;                  // 16 MB
  unsigned short* Vt  = Wbf + (size_t)N_TOKEN * N_EMBED;        // 16 MB

  int prep_blocks = (N_TOKEN * N_EMBED / 8) / NTHREADS;         // 4096
  prep_bf16<<<dim3(prep_blocks), dim3(NTHREADS), 0, stream>>>(lin_w, value_w, Wbf, Vt);

  int T = in_sizes[0] / N_EMBED;                                // 16384
  softmem_fused<<<dim3(T / BM), dim3(NTHREADS), 0, stream>>>(x, Wbf, lin_b, Vt, out);
}
